// MultiHeadAttention_33389075759788
// MI455X (gfx1250) — compile-verified, matrix operand order checked by reading, not run on hardware
//
#include <hip/hip_runtime.h>
#include <hip/hip_bf16.h>

typedef __attribute__((ext_vector_type(16))) __bf16 v16bf;
typedef __attribute__((ext_vector_type(8)))  float  v8f;
typedef __attribute__((ext_vector_type(4))) unsigned int u32x4;
typedef __attribute__((ext_vector_type(8))) int i32x8;
typedef __attribute__((ext_vector_type(4))) int i32x4;

#define WMMA_BF16(A, B, C) \
    __builtin_amdgcn_wmma_f32_16x16x32_bf16(false, (A), false, (B), (short)0, (C), false, false)

// Problem constants
#define BATCH   2
#define SEQ     2048
#define DMODEL  1024
#define NHEAD   16
#define DHEAD   64

// ---------------------------------------------------------------------------
// Tensor Data Mover support (gfx1250). Descriptor layout per
// CDNA5 ISA. Guarded so a toolchain without the
// builtin falls back to cooperative vector loads.
// ---------------------------------------------------------------------------
#if defined(__has_builtin)
#if __has_builtin(__builtin_amdgcn_tensor_load_to_lds) && \
    __has_builtin(__builtin_amdgcn_s_wait_tensorcnt)
#define USE_TDM 1
#endif
#endif
#ifndef USE_TDM
#define USE_TDM 0
#endif

#if USE_TDM
// 2D tile load: `rows` rows of `cols` bf16 elements, row stride `stride` elems,
// packed contiguously into LDS at byte offset `ldsOff`.
__device__ __forceinline__ void tdm_load_tile_bf16(unsigned ldsOff, const void* gptr,
                                                   unsigned rows, unsigned cols,
                                                   unsigned stride) {
    const unsigned long long ga = (unsigned long long)gptr;
    u32x4 g0;
    g0[0] = 1u;                                          // count=1, user mode
    g0[1] = ldsOff;                                      // lds_addr
    g0[2] = (unsigned)(ga & 0xffffffffu);                // global_addr[31:0]
    g0[3] = (unsigned)((ga >> 32) & 0x01ffffffu) | 0x80000000u; // addr[56:32] | type=2
    i32x8 g1;
    g1[0] = (int)(1u << 16);                             // data_size=1 (2 bytes)
    g1[1] = (int)((cols & 0xffffu) << 16);               // tensor_dim0[15:0] @bit48
    g1[2] = (int)(((cols >> 16) & 0xffffu) | ((rows & 0xffffu) << 16)); // dim0 hi | dim1 lo
    g1[3] = (int)(((rows >> 16) & 0xffffu) | ((cols & 0xffffu) << 16)); // dim1 hi | tile_dim0
    g1[4] = (int)(rows & 0xffffu);                       // tile_dim1 | tile_dim2=0
    g1[5] = (int)stride;                                 // tensor_dim0_stride[31:0]
    g1[6] = 0;                                           // stride hi | dim1_stride lo
    g1[7] = 0;
    const i32x4 z4 = {0, 0, 0, 0};
#if __has_include(<hip/amd_detail/amd_gfx1250_TDM.h>)
    const i32x8 z8 = {0, 0, 0, 0, 0, 0, 0, 0};
    __builtin_amdgcn_tensor_load_to_lds(g0, g1, z4, z4, z8, 0);
#else
    __builtin_amdgcn_tensor_load_to_lds(g0, g1, z4, z4, 0);
#endif
}
// Generic LDS pointer carries the LDS byte offset in its low 32 bits
// (ISA 10.2 aperture rules).
__device__ __forceinline__ unsigned lds_byte_off(const void* p) {
    return (unsigned)(unsigned long long)(size_t)p;
}
#endif

// ---------------------------------------------------------------------------
// Projection GEMM: Y = X @ W^T, X:[4096,1024] f32, W:[1024,1024] f32 (row = out
// feature), result scattered into [B,H,S,Dh] bf16.
// ---------------------------------------------------------------------------
__global__ __launch_bounds__(256) void mha_qkv_gemm(const float* __restrict__ X,
                                                    const float* __restrict__ W,
                                                    __bf16* __restrict__ Out) {
    __shared__ __bf16 As[128][36];
    __shared__ __bf16 Bs[128][36];

    const int t    = threadIdx.x;
    const int lane = t & 31;
    const int wv   = t >> 5;
    const int hi   = lane >> 4;
    const int l15  = lane & 15;
    const int wm   = wv & 3;
    const int wn   = wv >> 2;

    const int M0 = blockIdx.y * 128;
    const int N0 = blockIdx.x * 128;

    v8f acc[2][4] = {};

    const int lrow = t >> 3;
    const int lcol = (t & 7) * 4;

    for (int K0 = 0; K0 < DMODEL; K0 += 32) {
#pragma unroll
        for (int rr = 0; rr < 128; rr += 32) {
            const float* xp = X + (size_t)(M0 + lrow + rr) * DMODEL + K0 + lcol;
            const float* wp = W + (size_t)(N0 + lrow + rr) * DMODEL + K0 + lcol;
            const float4 av = *(const float4*)xp;
            As[lrow + rr][lcol + 0] = (__bf16)av.x;
            As[lrow + rr][lcol + 1] = (__bf16)av.y;
            As[lrow + rr][lcol + 2] = (__bf16)av.z;
            As[lrow + rr][lcol + 3] = (__bf16)av.w;
            const float4 bv = *(const float4*)wp;
            Bs[lrow + rr][lcol + 0] = (__bf16)bv.x;
            Bs[lrow + rr][lcol + 1] = (__bf16)bv.y;
            Bs[lrow + rr][lcol + 2] = (__bf16)bv.z;
            Bs[lrow + rr][lcol + 3] = (__bf16)bv.w;
            if (K0 + 32 < DMODEL) {
                __builtin_prefetch(xp + 32, 0, 1);
                __builtin_prefetch(wp + 32, 0, 1);
            }
        }
        __syncthreads();

        v16bf a[2], b[4];
#pragma unroll
        for (int mi = 0; mi < 2; ++mi) {
            const int m = wm * 32 + mi * 16 + l15;
#pragma unroll
            for (int j = 0; j < 8; ++j) {
                a[mi][j]     = As[m][hi * 8 + j];
                a[mi][8 + j] = As[m][16 + hi * 8 + j];
            }
        }
#pragma unroll
        for (int ni = 0; ni < 4; ++ni) {
            const int n = wn * 64 + ni * 16 + l15;
#pragma unroll
            for (int j = 0; j < 8; ++j) {
                b[ni][j]     = Bs[n][hi * 8 + j];
                b[ni][8 + j] = Bs[n][16 + hi * 8 + j];
            }
        }
#pragma unroll
        for (int mi = 0; mi < 2; ++mi)
#pragma unroll
            for (int ni = 0; ni < 4; ++ni)
                acc[mi][ni] = WMMA_BF16(a[mi], b[ni], acc[mi][ni]);
        __syncthreads();
    }

#pragma unroll
    for (int mi = 0; mi < 2; ++mi)
#pragma unroll
        for (int ni = 0; ni < 4; ++ni)
#pragma unroll
            for (int i = 0; i < 8; ++i) {
                const int r  = M0 + wm * 32 + mi * 16 + i + 8 * hi;
                const int o  = N0 + wn * 64 + ni * 16 + l15;
                const int bb = r >> 11;
                const int s  = r & (SEQ - 1);
                const int h  = o >> 6;
                const int dh = o & (DHEAD - 1);
                Out[(((size_t)(bb * NHEAD + h) * SEQ + s) * DHEAD) + dh] = (__bf16)acc[mi][ni][i];
            }
}

// ---------------------------------------------------------------------------
// Flash attention: grid = (S/128, B*H), block = 256 (8 waves).
// K/V tiles double-buffered in LDS; staged by the Tensor Data Mover (wave 0)
// with TENSORcnt tracking, falling back to cooperative loads if unavailable.
// ---------------------------------------------------------------------------
__global__ __launch_bounds__(256) void mha_flash_attn(const __bf16* __restrict__ Q,
                                                      const __bf16* __restrict__ K,
                                                      const __bf16* __restrict__ V,
                                                      const float* __restrict__ mask,
                                                      __bf16* __restrict__ Ctx) {
    __shared__ __bf16 Ks[2][64][64];
    __shared__ __bf16 Vs[2][64][64];
    __shared__ __bf16 Ps[8][16][64];

    const int t    = threadIdx.x;
    const int lane = t & 31;
    const int wv   = t >> 5;
    const int hi   = lane >> 4;
    const int l15  = lane & 15;

    const int bh = blockIdx.y;
    const int q0 = blockIdx.x * 128 + wv * 16;

    // Q fragments (A-matrix layout, two K=32 segments over Dh)
    v16bf aQ[2];
    {
        const __bf16* qr = Q + ((size_t)bh * SEQ + q0 + l15) * DHEAD;
#pragma unroll
        for (int seg = 0; seg < 2; ++seg)
#pragma unroll
            for (int j = 0; j < 8; ++j) {
                aQ[seg][j]     = qr[seg * 32 + hi * 8 + j];
                aQ[seg][8 + j] = qr[seg * 32 + 16 + hi * 8 + j];
            }
    }

    v8f O[4] = {};
    float mrun[8], lrun[8];
#pragma unroll
    for (int i = 0; i < 8; ++i) { mrun[i] = -3.0e38f; lrun[i] = 0.0f; }

    const float scale = 0.125f;

#if USE_TDM
    const __bf16* Kbh = K + (size_t)bh * SEQ * DHEAD;
    const __bf16* Vbh = V + (size_t)bh * SEQ * DHEAD;
    if (wv == 0) {
        tdm_load_tile_bf16(lds_byte_off(&Ks[0][0][0]), Kbh, 64, DHEAD, DHEAD);
        tdm_load_tile_bf16(lds_byte_off(&Vs[0][0][0]), Vbh, 64, DHEAD, DHEAD);
    }
#else
    const int srow = t >> 2;
    const int scol = (t & 3) * 16;
#endif

    int buf = 0;
    for (int kt = 0; kt < SEQ; kt += 64, buf ^= 1) {
#if USE_TDM
        if (wv == 0) {
            if (kt + 64 < SEQ) {
                // stream next tile into the other buffer, then wait for the
                // current tile (in-order TENSORcnt: 2 outstanding = next tile)
                tdm_load_tile_bf16(lds_byte_off(&Ks[buf ^ 1][0][0]),
                                   Kbh + (size_t)(kt + 64) * DHEAD, 64, DHEAD, DHEAD);
                tdm_load_tile_bf16(lds_byte_off(&Vs[buf ^ 1][0][0]),
                                   Vbh + (size_t)(kt + 64) * DHEAD, 64, DHEAD, DHEAD);
                __builtin_amdgcn_s_wait_tensorcnt(2);
            } else {
                __builtin_amdgcn_s_wait_tensorcnt(0);
            }
        }
        __syncthreads();
#else
        const __bf16* kp = K + ((size_t)bh * SEQ + kt + srow) * DHEAD + scol;
        const __bf16* vp = V + ((size_t)bh * SEQ + kt + srow) * DHEAD + scol;
        ((uint4*)&Ks[buf][srow][scol])[0] = ((const uint4*)kp)[0];
        ((uint4*)&Ks[buf][srow][scol])[1] = ((const uint4*)kp)[1];
        ((uint4*)&Vs[buf][srow][scol])[0] = ((const uint4*)vp)[0];
        ((uint4*)&Vs[buf][srow][scol])[1] = ((const uint4*)vp)[1];
        __syncthreads();
#endif

        // scores = Q K^T for 4 groups of 16 keys
        v8f sc[4] = {};
#pragma unroll
        for (int ni = 0; ni < 4; ++ni) {
            v16bf b0, b1;
            const int n = ni * 16 + l15;
#pragma unroll
            for (int j = 0; j < 8; ++j) {
                b0[j]     = Ks[buf][n][hi * 8 + j];
                b0[8 + j] = Ks[buf][n][16 + hi * 8 + j];
                b1[j]     = Ks[buf][n][32 + hi * 8 + j];
                b1[8 + j] = Ks[buf][n][48 + hi * 8 + j];
            }
            sc[ni] = WMMA_BF16(aQ[0], b0, sc[ni]);
            sc[ni] = WMMA_BF16(aQ[1], b1, sc[ni]);
        }

        // scale + additive mask; per-row max
        float pl[4][8];
        float mloc[8];
#pragma unroll
        for (int i = 0; i < 8; ++i) mloc[i] = -3.0e38f;
#pragma unroll
        for (int ni = 0; ni < 4; ++ni)
#pragma unroll
            for (int i = 0; i < 8; ++i) {
                const int qrow = q0 + i + 8 * hi;
                const int kcol = kt + ni * 16 + l15;
                const float s  = sc[ni][i] * scale + mask[(size_t)qrow * SEQ + kcol];
                pl[ni][i] = s;
                mloc[i]   = fmaxf(mloc[i], s);
            }
#pragma unroll
        for (int m = 1; m < 16; m <<= 1)
#pragma unroll
            for (int i = 0; i < 8; ++i)
                mloc[i] = fmaxf(mloc[i], __shfl_xor(mloc[i], m, 32));

        float alpha[8], lsum[8];
#pragma unroll
        for (int i = 0; i < 8; ++i) {
            const float mn = fmaxf(mrun[i], mloc[i]);
            alpha[i] = __expf(mrun[i] - mn);
            mrun[i]  = mn;
            lsum[i]  = 0.0f;
        }
#pragma unroll
        for (int ni = 0; ni < 4; ++ni)
#pragma unroll
            for (int i = 0; i < 8; ++i) {
                const float p = __expf(pl[ni][i] - mrun[i]);
                pl[ni][i] = p;
                lsum[i]  += p;
            }
#pragma unroll
        for (int m = 1; m < 16; m <<= 1)
#pragma unroll
            for (int i = 0; i < 8; ++i)
                lsum[i] += __shfl_xor(lsum[i], m, 32);
#pragma unroll
        for (int i = 0; i < 8; ++i) lrun[i] = lrun[i] * alpha[i] + lsum[i];

#pragma unroll
        for (int ni = 0; ni < 4; ++ni)
#pragma unroll
            for (int i = 0; i < 8; ++i)
                O[ni][i] *= alpha[i];

        // transpose P through LDS into A-fragment layout (per-wave region;
        // same-wave DS ops are ordered)
#pragma unroll
        for (int ni = 0; ni < 4; ++ni)
#pragma unroll
            for (int i = 0; i < 8; ++i)
                Ps[wv][i + 8 * hi][ni * 16 + l15] = (__bf16)pl[ni][i];

        v16bf aP[2];
        {
            const __bf16* pr = &Ps[wv][l15][0];
#pragma unroll
            for (int seg = 0; seg < 2; ++seg)
#pragma unroll
                for (int j = 0; j < 8; ++j) {
                    aP[seg][j]     = pr[seg * 32 + hi * 8 + j];
                    aP[seg][8 + j] = pr[seg * 32 + 16 + hi * 8 + j];
                }
        }

        // O += P @ V
#pragma unroll
        for (int ni = 0; ni < 4; ++ni) {
            v16bf b0, b1;
            const int n = ni * 16 + l15;
#pragma unroll
            for (int j = 0; j < 8; ++j) {
                b0[j]     = Vs[buf][hi * 8 + j][n];
                b0[8 + j] = Vs[buf][16 + hi * 8 + j][n];
                b1[j]     = Vs[buf][32 + hi * 8 + j][n];
                b1[8 + j] = Vs[buf][48 + hi * 8 + j][n];
            }
            O[ni] = WMMA_BF16(aP[0], b0, O[ni]);
            O[ni] = WMMA_BF16(aP[1], b1, O[ni]);
        }
        __syncthreads();
    }

    const int b = bh >> 4;
    const int h = bh & 15;
#pragma unroll
    for (int ni = 0; ni < 4; ++ni)
#pragma unroll
        for (int i = 0; i < 8; ++i) {
            const int qrow = q0 + i + 8 * hi;
            const float o  = O[ni][i] / lrun[i];
            Ctx[((size_t)b * SEQ + qrow) * DMODEL + h * DHEAD + ni * 16 + l15] = (__bf16)o;
        }
}

// ---------------------------------------------------------------------------
// Output projection: Out = Ctx(bf16) @ W0^T(f32->bf16), result fp32.
// ---------------------------------------------------------------------------
__global__ __launch_bounds__(256) void mha_out_gemm(const __bf16* __restrict__ A,
                                                    const float* __restrict__ W,
                                                    float* __restrict__ Out) {
    __shared__ __bf16 As[128][36];
    __shared__ __bf16 Bs[128][36];

    const int t    = threadIdx.x;
    const int lane = t & 31;
    const int wv   = t >> 5;
    const int hi   = lane >> 4;
    const int l15  = lane & 15;
    const int wm   = wv & 3;
    const int wn   = wv >> 2;

    const int M0 = blockIdx.y * 128;
    const int N0 = blockIdx.x * 128;

    v8f acc[2][4] = {};

    const int arow = t >> 1;
    const int acol = (t & 1) * 16;
    const int lrow = t >> 3;
    const int lcol = (t & 7) * 4;

    for (int K0 = 0; K0 < DMODEL; K0 += 32) {
        const __bf16* ap = A + (size_t)(M0 + arow) * DMODEL + K0 + acol;
        ((uint4*)&As[arow][acol])[0] = ((const uint4*)ap)[0];
        ((uint4*)&As[arow][acol])[1] = ((const uint4*)ap)[1];
#pragma unroll
        for (int rr = 0; rr < 128; rr += 32) {
            const float* wp = W + (size_t)(N0 + lrow + rr) * DMODEL + K0 + lcol;
            const float4 bvv = *(const float4*)wp;
            Bs[lrow + rr][lcol + 0] = (__bf16)bvv.x;
            Bs[lrow + rr][lcol + 1] = (__bf16)bvv.y;
            Bs[lrow + rr][lcol + 2] = (__bf16)bvv.z;
            Bs[lrow + rr][lcol + 3] = (__bf16)bvv.w;
        }
        __syncthreads();

        v16bf a[2], b[4];
#pragma unroll
        for (int mi = 0; mi < 2; ++mi) {
            const int m = wm * 32 + mi * 16 + l15;
#pragma unroll
            for (int j = 0; j < 8; ++j) {
                a[mi][j]     = As[m][hi * 8 + j];
                a[mi][8 + j] = As[m][16 + hi * 8 + j];
            }
        }
#pragma unroll
        for (int ni = 0; ni < 4; ++ni) {
            const int n = wn * 64 + ni * 16 + l15;
#pragma unroll
            for (int j = 0; j < 8; ++j) {
                b[ni][j]     = Bs[n][hi * 8 + j];
                b[ni][8 + j] = Bs[n][16 + hi * 8 + j];
            }
        }
#pragma unroll
        for (int mi = 0; mi < 2; ++mi)
#pragma unroll
            for (int ni = 0; ni < 4; ++ni)
                acc[mi][ni] = WMMA_BF16(a[mi], b[ni], acc[mi][ni]);
        __syncthreads();
    }

#pragma unroll
    for (int mi = 0; mi < 2; ++mi)
#pragma unroll
        for (int ni = 0; ni < 4; ++ni)
#pragma unroll
            for (int i = 0; i < 8; ++i) {
                const int r = M0 + wm * 32 + mi * 16 + i + 8 * hi;
                const int o = N0 + wn * 64 + ni * 16 + l15;
                Out[(size_t)r * DMODEL + o] = acc[mi][ni][i];
            }
}

// ---------------------------------------------------------------------------
extern "C" void kernel_launch(void* const* d_in, const int* in_sizes, int n_in,
                              void* d_out, int out_size, void* d_ws, size_t ws_size,
                              hipStream_t stream) {
    const float* qx   = (const float*)d_in[0];
    const float* kx   = (const float*)d_in[1];
    const float* vx   = (const float*)d_in[2];
    const float* mask = (const float*)d_in[3];
    const float* wq   = (const float*)d_in[4];
    const float* wk   = (const float*)d_in[5];
    const float* wvp  = (const float*)d_in[6];
    const float* w0   = (const float*)d_in[7];

    char* ws = (char*)d_ws;
    const size_t SEG = (size_t)8 * 1024 * 1024;
    __bf16* Q   = (__bf16*)(ws + 0 * SEG);
    __bf16* Kb  = (__bf16*)(ws + 1 * SEG);
    __bf16* Vb  = (__bf16*)(ws + 2 * SEG);
    __bf16* Ctx = (__bf16*)(ws + 3 * SEG);

    dim3 blk(256);
    dim3 gP(DMODEL / 128, (BATCH * SEQ) / 128);
    mha_qkv_gemm<<<gP, blk, 0, stream>>>(qx, wq, Q);
    mha_qkv_gemm<<<gP, blk, 0, stream>>>(kx, wk, Kb);
    mha_qkv_gemm<<<gP, blk, 0, stream>>>(vx, wvp, Vb);

    dim3 gA(SEQ / 128, BATCH * NHEAD);
    mha_flash_attn<<<gA, blk, 0, stream>>>(Q, Kb, Vb, mask, Ctx);

    mha_out_gemm<<<gP, blk, 0, stream>>>(Ctx, w0, (float*)d_out);
}
